// PointNetPPMvM_55456617726250
// MI455X (gfx1250) — compile-verified, matrix operand order checked by reading, not run on hardware
//
#include <hip/hip_runtime.h>
#include <hip/hip_bf16.h>
#include <math.h>

// MI455X / gfx1250: wave32, WMMA 16x16x32 f16->f32 for all MLP/FC GEMMs.
// <10 GFLOP, ~10 MB data -> latency bound. Fused group-gather->MLP->maxpool,
// one wave per group, maxpool straight from WMMA D fragments (shfl_xor 16).
// Feature gathers for SA2/SA3 use GLOBAL_LOAD_ASYNC_TO_LDS_B128 (ASYNCcnt path).

typedef __attribute__((ext_vector_type(16))) _Float16 v16h;
typedef __attribute__((ext_vector_type(8)))  float    v8f;

#define NBATCH 32
#define TEMP_INV (1.0f / 0.7f)
#define KAPPA_MAX 80.0f

// K-index map for 16-bit A/B fragments (wave32):
// lane L holds row/col (L&15); lanes<16 -> K {0..7,16..23}, lanes>=16 -> +8
__device__ __forceinline__ int kmap(int h, int hs) {
  return (h < 8 ? h : h + 8) + (hs ? 8 : 0);
}

// Async global->LDS copy of 16 bytes (per-lane addresses), tracked by ASYNCcnt.
__device__ __forceinline__ void async_g2l_b128(const void* gaddr, void* lds_generic) {
  unsigned lds32 = (unsigned)(uintptr_t)lds_generic;       // low 32 bits = LDS offset
  unsigned long long g64 = (unsigned long long)(uintptr_t)gaddr;
  asm volatile("global_load_async_to_lds_b128 %0, %1, off"
               :: "v"(lds32), "v"(g64)
               : "memory");
}
__device__ __forceinline__ void wait_asynccnt0() {
  asm volatile("s_wait_asynccnt 0x0" ::: "memory");
}

// act(32 x K) @ W(K x N) + b, ReLU, store f16 row-major into out (stride ldout).
__device__ __forceinline__ void wave_mlp_layer(
    const _Float16* in, int ldin, int K,
    const float* __restrict__ W, const float* __restrict__ bias, int N,
    _Float16* out, int ldout)
{
  const int lane = threadIdx.x & 31;
  const int col  = lane & 15;
  const int hs   = lane >> 4;
  for (int n0 = 0; n0 < N; n0 += 16) {
    v8f acc0 = {};
    v8f acc1 = {};
    for (int k0 = 0; k0 < K; k0 += 32) {
      v16h a0, a1, bf;
#pragma unroll
      for (int h = 0; h < 16; ++h) {
        int k = k0 + kmap(h, hs);
        a0[h] = in[col * ldin + k];          // rows 0..15 (LDS zero-padded)
        a1[h] = in[(col + 16) * ldin + k];   // rows 16..31
        bf[h] = (k < K) ? (_Float16)W[(size_t)k * N + n0 + col] : (_Float16)0.0f;
      }
      acc0 = __builtin_amdgcn_wmma_f32_16x16x32_f16(false, a0, false, bf,
                                                    (short)0, acc0, false, false);
      acc1 = __builtin_amdgcn_wmma_f32_16x16x32_f16(false, a1, false, bf,
                                                    (short)0, acc1, false, false);
    }
    float bv = bias[n0 + col];
#pragma unroll
    for (int r = 0; r < 8; ++r) {
      int m = r + hs * 8; // D layout: lanes<16 rows r, lanes>=16 rows r+8
      out[m * ldout + n0 + col]        = (_Float16)fmaxf(acc0[r] + bv, 0.0f);
      out[(m + 16) * ldout + n0 + col] = (_Float16)fmaxf(acc1[r] + bv, 0.0f);
    }
  }
}

// Last layer: ReLU then max over the 32 sample rows, written to global.
// Column n lives in lanes L and L^16 -> one shfl_xor(16) finishes the pool.
__device__ __forceinline__ void wave_mlp_layer_maxpool(
    const _Float16* in, int ldin, int K,
    const float* __restrict__ W, const float* __restrict__ bias, int N,
    float* __restrict__ gout)
{
  const int lane = threadIdx.x & 31;
  const int col  = lane & 15;
  const int hs   = lane >> 4;
  for (int n0 = 0; n0 < N; n0 += 16) {
    v8f acc0 = {};
    v8f acc1 = {};
    for (int k0 = 0; k0 < K; k0 += 32) {
      v16h a0, a1, bf;
#pragma unroll
      for (int h = 0; h < 16; ++h) {
        int k = k0 + kmap(h, hs);
        a0[h] = in[col * ldin + k];
        a1[h] = in[(col + 16) * ldin + k];
        bf[h] = (k < K) ? (_Float16)W[(size_t)k * N + n0 + col] : (_Float16)0.0f;
      }
      acc0 = __builtin_amdgcn_wmma_f32_16x16x32_f16(false, a0, false, bf,
                                                    (short)0, acc0, false, false);
      acc1 = __builtin_amdgcn_wmma_f32_16x16x32_f16(false, a1, false, bf,
                                                    (short)0, acc1, false, false);
    }
    float bv = bias[n0 + col];
    float mx = 0.0f; // ReLU floor == 0 -> start at 0 == max over relu'd rows
#pragma unroll
    for (int r = 0; r < 8; ++r)
      mx = fmaxf(mx, fmaxf(acc0[r] + bv, acc1[r] + bv));
    mx = fmaxf(mx, __shfl_xor(mx, 16, 32));
    if (lane < 16) gout[n0 + col] = mx;
  }
}

// Fused: gather group rows (async G->LDS for feature rows) -> 3-layer MLP
// (WMMA) -> maxpool over 32 samples. One wave (32 threads) per (batch, centroid).
template <int KPADMAX, int CFMAX>
__global__ void __launch_bounds__(32) sa_kernel(
    const float* __restrict__ src_xyz, int Nsrc,
    const float* __restrict__ src_feats, int Cf,
    const int* __restrict__ nidx, const float* __restrict__ new_xyz,
    int M, int S,
    const float* __restrict__ w0, const float* __restrict__ b0, int N0,
    const float* __restrict__ w1, const float* __restrict__ b1, int N1,
    const float* __restrict__ w2, const float* __restrict__ b2, int N2,
    float* __restrict__ out_feats, int group_all)
{
  __shared__ _Float16 buf0[32 * KPADMAX];
  __shared__ _Float16 buf1[32 * KPADMAX];
  __shared__ __align__(16) float stage[32 * CFMAX];
  const int bm = blockIdx.x;
  const int b  = bm / M;
  const int s  = threadIdx.x; // sample 0..31
  const int K0 = 3 + Cf;
  const int K0pad = (K0 + 31) & ~31;

  __builtin_prefetch(w0, 0, 1);
  __builtin_prefetch(w1, 0, 1);
  __builtin_prefetch(w2, 0, 1);

  int idx;
  float cx = 0.f, cy = 0.f, cz = 0.f;
  if (group_all) {
    idx = s;
  } else {
    idx = nidx[(size_t)bm * S + s];
    cx = new_xyz[bm * 3 + 0];
    cy = new_xyz[bm * 3 + 1];
    cz = new_xyz[bm * 3 + 2];
  }

  // Kick off async feature-row gather first (ASYNCcnt), overlap with xyz work.
  float* drow = stage + s * CFMAX;
  if (Cf > 0) {
    const float* f = src_feats + ((size_t)b * Nsrc + idx) * Cf;
    for (int c = 0; c < Cf; c += 4)       // Cf is a multiple of 4 -> 16B chunks
      async_g2l_b128(f + c, drow + c);
  }

  const float* p = src_xyz + ((size_t)b * Nsrc + idx) * 3;
  _Float16* row = buf0 + s * KPADMAX;
  row[0] = (_Float16)(p[0] - cx);
  row[1] = (_Float16)(p[1] - cy);
  row[2] = (_Float16)(p[2] - cz);
  for (int k = K0; k < K0pad; ++k) row[k] = (_Float16)0.0f;

  if (Cf > 0) {
    wait_asynccnt0();
    for (int c = 0; c < Cf; ++c) row[3 + c] = (_Float16)drow[c];
  }
  __syncthreads();

  wave_mlp_layer(buf0, KPADMAX, K0, w0, b0, N0, buf1, KPADMAX);
  __syncthreads();
  wave_mlp_layer(buf1, KPADMAX, N0, w1, b1, N1, buf0, KPADMAX);
  __syncthreads();
  wave_mlp_layer_maxpool(buf0, KPADMAX, N1, w2, b2, N2,
                         out_feats + (size_t)bm * N2);
}

// Farthest point sampling: one block per batch; running min-dist array in LDS
// (64 KB for N=16384 -- only feasible thanks to CDNA5's 320 KB LDS).
template <int NMAX>
__global__ void fps_kernel(const float* __restrict__ xyz, int N, int npoint,
                           int* __restrict__ fidx)
{
  __shared__ float sdist[NMAX];
  __shared__ float rmax[256];
  __shared__ int   rarg[256];
  __shared__ int   sfar;
  const int b = blockIdx.x, t = threadIdx.x;
  const float* base = xyz + (size_t)b * N * 3;
  for (int i = t; i < N; i += 256) sdist[i] = 1e10f;
  if (t == 0) sfar = 0;
  __syncthreads();
  for (int it = 0; it < npoint; ++it) {
    const int far = sfar;
    if (t == 0) fidx[b * npoint + it] = far;
    const float cx = base[far * 3 + 0], cy = base[far * 3 + 1], cz = base[far * 3 + 2];
    float lmax = -1.0f; int larg = 0;
    for (int i = t; i < N; i += 256) {
      float dx = base[i * 3 + 0] - cx;
      float dy = base[i * 3 + 1] - cy;
      float dz = base[i * 3 + 2] - cz;
      float d  = dx * dx + dy * dy + dz * dz;
      float nd = fminf(sdist[i], d);
      sdist[i] = nd;
      if (nd > lmax) { lmax = nd; larg = i; }
    }
    rmax[t] = lmax; rarg[t] = larg;
    __syncthreads();
    for (int off = 128; off > 0; off >>= 1) {
      if (t < off) {
        float o = rmax[t + off]; int oa = rarg[t + off];
        if (o > rmax[t] || (o == rmax[t] && oa < rarg[t])) { rmax[t] = o; rarg[t] = oa; }
      }
      __syncthreads();
    }
    if (t == 0) sfar = rarg[0];
    __syncthreads();
  }
}

__global__ void gather3_kernel(const float* __restrict__ src, int N,
                               const int* __restrict__ idx, int M,
                               float* __restrict__ dst)
{
  int i = blockIdx.x * blockDim.x + threadIdx.x;
  if (i >= NBATCH * M) return;
  int b = i / M;
  int id = idx[i];
  const float* p = src + ((size_t)b * N + id) * 3;
  dst[i * 3 + 0] = p[0];
  dst[i * 3 + 1] = p[1];
  dst[i * 3 + 2] = p[2];
}

// top-S nearest via S lexicographic-min passes (matches top_k(-d) order).
__global__ void knn_kernel(const float* __restrict__ src, int N,
                           const float* __restrict__ q, int M, int S,
                           int* __restrict__ nidx)
{
  __shared__ float rbest[256];
  __shared__ int   rarg[256];
  __shared__ float s_ld;
  __shared__ int   s_li;
  const int bm = blockIdx.x, b = bm / M, t = threadIdx.x;
  const float cx = q[bm * 3 + 0], cy = q[bm * 3 + 1], cz = q[bm * 3 + 2];
  const float* base = src + (size_t)b * N * 3;
  if (t == 0) { s_ld = -1e30f; s_li = -1; }
  __syncthreads();
  for (int sel = 0; sel < S; ++sel) {
    const float ld = s_ld; const int li = s_li;
    float best = 3e38f; int barg = 0x7fffffff;
    for (int i = t; i < N; i += 256) {
      float dx = base[i * 3 + 0] - cx;
      float dy = base[i * 3 + 1] - cy;
      float dz = base[i * 3 + 2] - cz;
      float d  = dx * dx + dy * dy + dz * dz;
      bool after = (d > ld) || (d == ld && i > li);
      if (after && (d < best || (d == best && i < barg))) { best = d; barg = i; }
    }
    rbest[t] = best; rarg[t] = barg;
    __syncthreads();
    for (int off = 128; off > 0; off >>= 1) {
      if (t < off) {
        float o = rbest[t + off]; int oa = rarg[t + off];
        if (o < rbest[t] || (o == rbest[t] && oa < rarg[t])) { rbest[t] = o; rarg[t] = oa; }
      }
      __syncthreads();
    }
    if (t == 0) { nidx[(size_t)bm * S + sel] = rarg[0]; s_ld = rbest[0]; s_li = rarg[0]; }
    __syncthreads();
  }
}

// Y(32 x N) = X(32 x K) @ W(K x N) + b ; one wave per 16-col tile, WMMA.
__global__ void __launch_bounds__(32) fc_kernel(
    const float* __restrict__ X, const float* __restrict__ W,
    const float* __restrict__ bias, int K, int N, float* __restrict__ Y)
{
  const int n0   = blockIdx.x * 16;
  const int lane = threadIdx.x & 31;
  const int col  = lane & 15;
  const int hs   = lane >> 4;
  __builtin_prefetch(W, 0, 1);
  v8f acc0 = {};
  v8f acc1 = {};
  for (int k0 = 0; k0 < K; k0 += 32) {
    v16h a0, a1, bf;
#pragma unroll
    for (int h = 0; h < 16; ++h) {
      int k = k0 + kmap(h, hs);
      a0[h] = (_Float16)X[(size_t)col * K + k];
      a1[h] = (_Float16)X[(size_t)(col + 16) * K + k];
      bf[h] = (_Float16)W[(size_t)k * N + n0 + col];
    }
    acc0 = __builtin_amdgcn_wmma_f32_16x16x32_f16(false, a0, false, bf,
                                                  (short)0, acc0, false, false);
    acc1 = __builtin_amdgcn_wmma_f32_16x16x32_f16(false, a1, false, bf,
                                                  (short)0, acc1, false, false);
  }
  float bv = bias[n0 + col];
#pragma unroll
  for (int r = 0; r < 8; ++r) {
    int m = r + hs * 8;
    Y[(size_t)m * N + n0 + col]        = acc0[r] + bv;
    Y[(size_t)(m + 16) * N + n0 + col] = acc1[r] + bv;
  }
}

__global__ void ln_relu_kernel(float* __restrict__ x, const float* __restrict__ g,
                               const float* __restrict__ bb, int D)
{
  __shared__ float ssum[256], ssq[256];
  const int b = blockIdx.x, t = threadIdx.x;
  float s = 0.f, sq = 0.f;
  for (int i = t; i < D; i += 256) {
    float v = x[(size_t)b * D + i];
    s += v; sq += v * v;
  }
  ssum[t] = s; ssq[t] = sq;
  __syncthreads();
  for (int off = 128; off > 0; off >>= 1) {
    if (t < off) { ssum[t] += ssum[t + off]; ssq[t] += ssq[t + off]; }
    __syncthreads();
  }
  const float mean = ssum[0] / D;
  const float var  = ssq[0] / D - mean * mean;
  const float inv  = rsqrtf(var + 1e-5f);
  for (int i = t; i < D; i += 256) {
    float v = (x[(size_t)b * D + i] - mean) * inv * g[i] + bb[i];
    x[(size_t)b * D + i] = fmaxf(v, 0.0f);
  }
}

// pi/mu/kappa heads + softmax/atan2/softplus; one block per batch row.
__global__ void head_kernel(const float* __restrict__ X,
                            const float* __restrict__ pi_w, const float* __restrict__ pi_b,
                            const float* __restrict__ mu_w, const float* __restrict__ mu_b,
                            const float* __restrict__ kp_w, const float* __restrict__ kp_b,
                            float* __restrict__ out)
{
  __shared__ float xr[256];
  __shared__ float logit[4], mus[8], kapz[4];
  const int b = blockIdx.x, t = threadIdx.x;
  for (int i = t; i < 256; i += 64) xr[i] = X[(size_t)b * 256 + i];
  __syncthreads();
  if (t < 4) {
    float a = pi_b[t];
    for (int i = 0; i < 256; ++i) a += xr[i] * pi_w[i * 4 + t];
    logit[t] = a;
  } else if (t < 12) {
    int j = t - 4;
    float a = mu_b[j];
    for (int i = 0; i < 256; ++i) a += xr[i] * mu_w[i * 8 + j];
    mus[j] = a;
  } else if (t < 16) {
    int j = t - 12;
    float a = kp_b[j];
    for (int i = 0; i < 256; ++i) a += xr[i] * kp_w[i * 4 + j];
    kapz[j] = a;
  }
  __syncthreads();
  if (t < 4) {
    float mx = fmaxf(fmaxf(logit[0], logit[1]), fmaxf(logit[2], logit[3])) * TEMP_INV;
    float sum = 0.f;
    for (int j = 0; j < 4; ++j) sum += expf(logit[j] * TEMP_INV - mx);
    float wgt = expf(logit[t] * TEMP_INV - mx) / sum;
    float c = mus[2 * t], sn = mus[2 * t + 1];
    float nrm = sqrtf(c * c + sn * sn);
    float d = fmaxf(nrm, 1e-4f);
    float cu = c / d, su = sn / d;
    float n2 = sqrtf(cu * cu + su * su);
    if (n2 < 1e-3f) { cu = 1.0f; su = 0.0f; }
    float mu = atan2f(su, cu);
    float z = kapz[t];
    float sp = (z > 20.0f) ? z : log1pf(expf(z));
    float kap = fminf(sp + 1e-6f, KAPPA_MAX);
    out[b * 4 + t]       = mu;   // mu    (32x4)
    out[128 + b * 4 + t] = kap;  // kappa (32x4)
    out[256 + b * 4 + t] = wgt;  // weight(32x4)
  }
}

extern "C" void kernel_launch(void* const* d_in, const int* in_sizes, int n_in,
                              void* d_out, int out_size, void* d_ws, size_t ws_size,
                              hipStream_t stream)
{
  (void)in_sizes; (void)n_in; (void)out_size; (void)ws_size;
  const float* xyz = (const float*)d_in[0];
  const float* sa1w0 = (const float*)d_in[1];  const float* sa1b0 = (const float*)d_in[2];
  const float* sa1w1 = (const float*)d_in[3];  const float* sa1b1 = (const float*)d_in[4];
  const float* sa1w2 = (const float*)d_in[5];  const float* sa1b2 = (const float*)d_in[6];
  const float* sa2w0 = (const float*)d_in[7];  const float* sa2b0 = (const float*)d_in[8];
  const float* sa2w1 = (const float*)d_in[9];  const float* sa2b1 = (const float*)d_in[10];
  const float* sa2w2 = (const float*)d_in[11]; const float* sa2b2 = (const float*)d_in[12];
  const float* sa3w0 = (const float*)d_in[13]; const float* sa3b0 = (const float*)d_in[14];
  const float* sa3w1 = (const float*)d_in[15]; const float* sa3b1 = (const float*)d_in[16];
  const float* sa3w2 = (const float*)d_in[17]; const float* sa3b2 = (const float*)d_in[18];
  const float* fc1w = (const float*)d_in[19]; const float* fc1b = (const float*)d_in[20];
  const float* ln1g = (const float*)d_in[21]; const float* ln1b = (const float*)d_in[22];
  const float* fc2w = (const float*)d_in[23]; const float* fc2b = (const float*)d_in[24];
  const float* ln2g = (const float*)d_in[25]; const float* ln2b = (const float*)d_in[26];
  const float* piw  = (const float*)d_in[27]; const float* pib  = (const float*)d_in[28];
  const float* muw  = (const float*)d_in[29]; const float* mub  = (const float*)d_in[30];
  const float* kpw  = (const float*)d_in[31]; const float* kpb  = (const float*)d_in[32];

  char* wsp = (char*)d_ws;
  auto alloc = [&](size_t bytes) -> void* {
    void* p = (void*)wsp;
    wsp += (bytes + 255) & ~((size_t)255);
    return p;
  };
  int*   fidx1 = (int*)  alloc((size_t)32 * 128 * 4);
  float* nx1   = (float*)alloc((size_t)32 * 128 * 3 * 4);
  int*   nidx1 = (int*)  alloc((size_t)32 * 128 * 32 * 4);
  float* l1f   = (float*)alloc((size_t)32 * 128 * 128 * 4);
  int*   fidx2 = (int*)  alloc((size_t)32 * 32 * 4);
  float* nx2   = (float*)alloc((size_t)32 * 32 * 3 * 4);
  int*   nidx2 = (int*)  alloc((size_t)32 * 32 * 32 * 4);
  float* l2f   = (float*)alloc((size_t)32 * 32 * 256 * 4);
  float* l3f   = (float*)alloc((size_t)32 * 1024 * 4);
  float* y1    = (float*)alloc((size_t)32 * 512 * 4);
  float* y2    = (float*)alloc((size_t)32 * 256 * 4);

  // ---- SA1: 16384 -> 128 centroids, 32-NN, MLP 3->64->64->128, maxpool ----
  fps_kernel<16384><<<32, 256, 0, stream>>>(xyz, 16384, 128, fidx1);
  gather3_kernel<<<(32 * 128 + 255) / 256, 256, 0, stream>>>(xyz, 16384, fidx1, 128, nx1);
  knn_kernel<<<32 * 128, 256, 0, stream>>>(xyz, 16384, nx1, 128, 32, nidx1);
  sa_kernel<64, 4><<<32 * 128, 32, 0, stream>>>(
      xyz, 16384, nullptr, 0, nidx1, nx1, 128, 32,
      sa1w0, sa1b0, 64, sa1w1, sa1b1, 64, sa1w2, sa1b2, 128, l1f, 0);

  // ---- SA2: 128 -> 32 centroids, 32-NN, MLP 131->128->128->256, maxpool ----
  fps_kernel<128><<<32, 256, 0, stream>>>(nx1, 128, 32, fidx2);
  gather3_kernel<<<(32 * 32 + 255) / 256, 256, 0, stream>>>(nx1, 128, fidx2, 32, nx2);
  knn_kernel<<<32 * 32, 256, 0, stream>>>(nx1, 128, nx2, 32, 32, nidx2);
  sa_kernel<160, 128><<<32 * 32, 32, 0, stream>>>(
      nx1, 128, l1f, 128, nidx2, nx2, 32, 32,
      sa2w0, sa2b0, 128, sa2w1, sa2b1, 128, sa2w2, sa2b2, 256, l2f, 0);

  // ---- SA3: group_all over 32 points, MLP 259->256->512->1024, maxpool ----
  sa_kernel<512, 256><<<32, 32, 0, stream>>>(
      nx2, 32, l2f, 256, nullptr, nullptr, 1, 32,
      sa3w0, sa3b0, 256, sa3w1, sa3b1, 512, sa3w2, sa3b2, 1024, l3f, 1);

  // ---- FC head ----
  fc_kernel<<<512 / 16, 32, 0, stream>>>(l3f, fc1w, fc1b, 1024, 512, y1);
  ln_relu_kernel<<<32, 256, 0, stream>>>(y1, ln1g, ln1b, 512);
  fc_kernel<<<256 / 16, 32, 0, stream>>>(y1, fc2w, fc2b, 512, 256, y2);
  ln_relu_kernel<<<32, 256, 0, stream>>>(y2, ln2g, ln2b, 256);
  head_kernel<<<32, 64, 0, stream>>>(y2, piw, pib, muw, mub, kpw, kpb, (float*)d_out);
}
